// HybridDecoderLayers_16080357556390
// MI455X (gfx1250) — matrix operand order repaired and checked by reading, not compiled, not run on hardware
//
#include <hip/hip_runtime.h>

// ---------------------------------------------------------------------------
// CDNA5 (gfx1250) hybrid Mamba2 + MLP decoder layer.
// GEMMs: v_wmma_f32_16x16x32_bf16 (bf16 in, f32 acc); A tiles via TDM
// (tensor_load_to_lds, TENSORcnt) -- the MI455X async data-mover path.
// ---------------------------------------------------------------------------

typedef __attribute__((ext_vector_type(16))) __bf16 v16bf;
typedef __attribute__((ext_vector_type(8)))  __bf16 v8bf;
typedef __attribute__((ext_vector_type(8)))  float  v8f;
typedef unsigned int u32x4 __attribute__((ext_vector_type(4)));
typedef int          i32x8 __attribute__((ext_vector_type(8)));
typedef int          i32x4 __attribute__((ext_vector_type(4)));

__device__ __forceinline__ __bf16 tobf(float f) { return (__bf16)f; }
__device__ __forceinline__ float siluf(float x) { return x / (1.f + __expf(-x)); }

__device__ __forceinline__ v8f vzero8() {
  v8f z;
#pragma unroll
  for (int i = 0; i < 8; ++i) z[i] = 0.f;
  return z;
}

// A fragment: 16(M) x 32(K) bf16, source row-major [M][K] bf16, stride lda.
__device__ __forceinline__ v16bf frag_a(const __bf16* p0, int lda) {
  int lane = threadIdx.x & 31;
  const __bf16* r = p0 + (lane & 15) * lda + ((lane >> 4) << 3);
  v8bf lo = *(const v8bf*)(r);
  v8bf hi = *(const v8bf*)(r + 16);
  return __builtin_shufflevector(lo, hi, 0, 1, 2, 3, 4, 5, 6, 7,
                                 8, 9, 10, 11, 12, 13, 14, 15);
}

// A fragment built from an f32 LDS tile (TDM-delivered), converting to bf16.
__device__ __forceinline__ v16bf frag_a_f32(const float* p0, int lda) {
  int lane = threadIdx.x & 31;
  const float* r = p0 + (lane & 15) * lda + ((lane >> 4) << 3);
  v16bf f;
#pragma unroll
  for (int i = 0; i < 8; ++i) { f[i] = tobf(r[i]); f[i + 8] = tobf(r[16 + i]); }
  return f;
}

// B fragment: 32(K) x 16(N) bf16, source stored [N][K]-major, stride ldb.
__device__ __forceinline__ v16bf frag_b(const __bf16* p0, int ldb) {
  int lane = threadIdx.x & 31;
  const __bf16* r = p0 + (lane & 15) * ldb + ((lane >> 4) << 3);
  v8bf lo = *(const v8bf*)(r);
  v8bf hi = *(const v8bf*)(r + 16);
  return __builtin_shufflevector(lo, hi, 0, 1, 2, 3, 4, 5, 6, 7,
                                 8, 9, 10, 11, 12, 13, 14, 15);
}

__device__ __forceinline__ v8f wmma_bf16(v16bf a, v16bf b, v8f c) {
  return __builtin_amdgcn_wmma_f32_16x16x32_bf16(false, a, false, b,
                                                 (short)0, c, false, false);
}

// ---------------------------------------------------------------------------
// TDM: DMA a 2D f32 tile [tile_y][tile_x] (row-major, row stride
// stride_elems) from global memory into LDS at lds_off. D# layout per
// CDNA5 ISA ch.8 (group0 128b, group1 256b, groups 2/3 zero for 2D).
// Toolchain uses the 6-arg builtin form: (g0, g1, g2, g3, g4, cpol).
// ---------------------------------------------------------------------------
__device__ __forceinline__ void tdm_load_2d_f32(
    unsigned lds_off, const float* gptr, unsigned tile_x, unsigned tile_y,
    unsigned stride_elems, unsigned tensor_x, unsigned tensor_y) {
  unsigned long long ga = (unsigned long long)(__SIZE_TYPE__)gptr;
  u32x4 g0;
  g0[0] = 1u;                                    // count=1 (valid), user mode
  g0[1] = lds_off;                               // lds_addr (bytes)
  g0[2] = (unsigned)(ga & 0xffffffffull);        // global_addr[31:0]
  g0[3] = (unsigned)((ga >> 32) & 0x01fffffful)  // global_addr[56:32]
          | (2u << 30);                          // type=2 ("image")
  i32x8 g1;
  g1[0] = (int)(2u << 16);                       // data_size=2 (4 bytes)
  g1[1] = (int)((tensor_x & 0xffffu) << 16);     // tensor_dim0 low 16
  g1[2] = (int)((tensor_x >> 16) | ((tensor_y & 0xffffu) << 16));
  g1[3] = (int)((tensor_y >> 16) | (tile_x << 16));  // tile_dim0
  g1[4] = (int)(tile_y & 0xffffu);               // tile_dim1 (tile_dim2=0)
  g1[5] = (int)stride_elems;                     // tensor_dim0_stride[31:0]
  g1[6] = 0;                                     // stride hi / dim1_stride
  g1[7] = 0;
  i32x4 gz;
  gz[0] = 0; gz[1] = 0; gz[2] = 0; gz[3] = 0;
  i32x8 gz8;
#pragma unroll
  for (int i = 0; i < 8; ++i) gz8[i] = 0;
  __builtin_amdgcn_tensor_load_to_lds(g0, g1, gz, gz, gz8, 0);
}

__device__ __forceinline__ unsigned lds_offset_of(const void* p) {
  return (unsigned)(__SIZE_TYPE__)p;  // flat LDS addr[31:0] == LDS offset
}

// ---------------------------------------------------------------------------
// RMSNorm: one block per row.
// ---------------------------------------------------------------------------
__global__ __launch_bounds__(256) void rmsnorm_kernel(
    const float* __restrict__ in, const float* __restrict__ w,
    float* __restrict__ out, int cols) {
  __shared__ float red[8];
  int row = blockIdx.x;
  const float* x = in + (size_t)row * cols;
  float ss = 0.f;
  for (int i = threadIdx.x; i < cols; i += 256) { float v = x[i]; ss += v * v; }
#pragma unroll
  for (int o = 16; o > 0; o >>= 1) ss += __shfl_xor(ss, o, 32);
  if ((threadIdx.x & 31) == 0) red[threadIdx.x >> 5] = ss;
  __syncthreads();
  ss = 0.f;
#pragma unroll
  for (int i = 0; i < 8; ++i) ss += red[i];
  float scale = rsqrtf(ss / (float)cols + 1e-5f);
  for (int i = threadIdx.x; i < cols; i += 256)
    out[(size_t)row * cols + i] = x[i] * scale * w[i];
}

// Gated norm: out = rmsnorm(y * silu(z)) * norm_w, cols = 4096, z from zxbcdt.
__global__ __launch_bounds__(256) void gated_norm_kernel(
    const float* __restrict__ y, const float* __restrict__ zx,
    const float* __restrict__ nw, float* __restrict__ out) {
  __shared__ float sv[4096];
  __shared__ float red[8];
  int row = blockIdx.x;
  float ss = 0.f;
  for (int i = threadIdx.x; i < 4096; i += 256) {
    float z = zx[(size_t)row * 10304 + i];
    float g = y[(size_t)row * 4096 + i] * siluf(z);
    sv[i] = g;
    ss += g * g;
  }
#pragma unroll
  for (int o = 16; o > 0; o >>= 1) ss += __shfl_xor(ss, o, 32);
  if ((threadIdx.x & 31) == 0) red[threadIdx.x >> 5] = ss;
  __syncthreads();
  ss = 0.f;
#pragma unroll
  for (int i = 0; i < 8; ++i) ss += red[i];
  float scale = rsqrtf(ss / 4096.f + 1e-5f);
  for (int i = threadIdx.x; i < 4096; i += 256)
    out[(size_t)row * 4096 + i] = sv[i] * scale * nw[i];
}

// ---------------------------------------------------------------------------
// dt = softplus(zx[:,10240+h] + dt_bias[h])
// ---------------------------------------------------------------------------
__global__ __launch_bounds__(256) void dt_kernel(
    const float* __restrict__ zx, const float* __restrict__ dt_bias,
    float* __restrict__ dt) {
  int i = blockIdx.x * 256 + threadIdx.x;  // 4096*64
  int row = i >> 6, h = i & 63;
  float v = zx[(size_t)row * 10304 + 10240 + h] + dt_bias[h];
  dt[i] = (v > 20.f) ? v : log1pf(__expf(v));
}

// ---------------------------------------------------------------------------
// Causal depthwise conv (k=4) over [x,B,C] cols of zxbcdt, + bias, SiLU.
// ---------------------------------------------------------------------------
__global__ __launch_bounds__(256) void conv_kernel(
    const float* __restrict__ zx, const float* __restrict__ cw,
    const float* __restrict__ cb, float* __restrict__ xBC) {
  int c = blockIdx.x * 256 + threadIdx.x;  // 0..6143
  int row = blockIdx.y;                    // 0..4095
  int t = row & 2047;
  float acc = cb[c];
#pragma unroll
  for (int j = 0; j < 4; ++j) {
    int tt = t - 3 + j;
    if (tt >= 0)
      acc += zx[(size_t)(row - t + tt) * 10304 + 4096 + c] * cw[c * 4 + j];
  }
  xBC[(size_t)row * 6144 + c] = siluf(acc);
}

// ---------------------------------------------------------------------------
// WMMA GEMM: C[M,N] = A[M,K] @ W[K,N]. Block tile 128x128x32, 8 waves.
// A tile: TDM DMA global->LDS (f32), converted to bf16 at fragment build.
// W tile: manual transposed staging (TDM cannot transpose) to [n][k] bf16.
// EPI: 0 = store, 1 = acc+aux, 2 = silu(acc), 3 = acc*aux
// ---------------------------------------------------------------------------
template <int EPI>
__global__ __launch_bounds__(256) void gemm_kernel(
    const float* __restrict__ A, const float* __restrict__ W,
    float* __restrict__ C, const float* __restrict__ aux,
    int M, int N, int K) {
  __shared__ float  sAf[128 * 32];  // [m][k] f32, TDM-filled (16KB)
  __shared__ __bf16 sB[128 * 32];   // [n][k] bf16 (8KB)
  int tid = threadIdx.x;
  int lane = tid & 31, wid = tid >> 5;
  int wm = wid >> 1, wn = wid & 1;
  int m0 = blockIdx.y * 128;
  int n0 = blockIdx.x * 128;
  unsigned sAoff = lds_offset_of(sAf);

  v8f acc[2][4];
#pragma unroll
  for (int i = 0; i < 2; ++i)
#pragma unroll
    for (int j = 0; j < 4; ++j) acc[i][j] = vzero8();

  for (int k0 = 0; k0 < K; k0 += 32) {
    // Wave 0: kick off TDM for the 128x32 f32 A tile (rows m0.., cols k0..).
    if (wid == 0) {
      tdm_load_2d_f32(sAoff, A + (size_t)m0 * K + k0,
                      /*tile_x=*/32, /*tile_y=*/128,
                      /*stride=*/(unsigned)K,
                      /*tensor_x=*/(unsigned)K, /*tensor_y=*/(unsigned)M);
    }
    // All waves: stage W tile transposed to [n][k] bf16 (overlaps with DMA).
#pragma unroll
    for (int i = 0; i < 16; ++i) {
      int idx = tid + i * 256;
      int k = idx >> 7, n = idx & 127;
      float v = 0.f;
      if (n0 + n < N) v = W[(size_t)(k0 + k) * N + n0 + n];
      sB[n * 32 + k] = tobf(v);
    }
    if (k0 + 32 < K)  // gfx1250 global_prefetch_b8 on next W tile
      __builtin_prefetch(&W[(size_t)(k0 + 32 + (tid & 31)) * N + n0], 0, 1);
    if (wid == 0) __builtin_amdgcn_s_wait_tensorcnt(0);
    __syncthreads();

#pragma unroll
    for (int i = 0; i < 2; ++i) {
      v16bf a = frag_a_f32(sAf + (wm * 32 + i * 16) * 32, 32);
#pragma unroll
      for (int j = 0; j < 4; ++j) {
        v16bf b = frag_b(sB + (wn * 64 + j * 16) * 32, 32);
        acc[i][j] = wmma_bf16(a, b, acc[i][j]);
      }
    }
    __syncthreads();
  }

  int rlo = (lane >> 4) << 3;
  int col = lane & 15;
#pragma unroll
  for (int i = 0; i < 2; ++i)
#pragma unroll
    for (int j = 0; j < 4; ++j) {
      int mb = m0 + wm * 32 + i * 16 + rlo;
      int nb = n0 + wn * 64 + j * 16 + col;
      if (nb < N) {
#pragma unroll
        for (int v = 0; v < 8; ++v) {
          size_t idx = (size_t)(mb + v) * N + nb;
          float val = acc[i][j][v];
          if (EPI == 1) val += aux[idx];
          else if (EPI == 2) val = siluf(val);
          else if (EPI == 3) val *= aux[idx];
          C[idx] = val;
        }
      }
    }
}

// ---------------------------------------------------------------------------
// SSD-A: per (chunk c, b*h). Y_diag, per-chunk states, chunk decay, dAc.
// G = C @ B^T and Y_diag = (G o L) @ xdt run on WMMA.
// Dynamic LDS: sXT[64][128] | 32KB region (sC+sB -> sG) | sdt[128] | sdAc[128]
// ---------------------------------------------------------------------------
__global__ __launch_bounds__(256) void ssd_a_kernel(
    const float* __restrict__ xBC, const float* __restrict__ dtbuf,
    const float* __restrict__ A_log, float* __restrict__ y,
    float* __restrict__ states, float* __restrict__ cdecay,
    float* __restrict__ dAc_out) {
  extern __shared__ char smem[];
  __bf16* sXT = (__bf16*)smem;                    // [64][128]  x*dt transposed
  __bf16* sC  = (__bf16*)(smem + 16384);          // [128][64]
  __bf16* sB  = (__bf16*)(smem + 16384 + 16384);  // [128][64]
  __bf16* sG  = (__bf16*)(smem + 16384);          // [128][128] overlays sC/sB
  float* sdt  = (float*)(smem + 49152);
  float* sdAc = (float*)(smem + 49152 + 512);

  int cIdx = blockIdx.x;  // chunk
  int bh = blockIdx.y;    // b*64 + h
  int h = bh & 63, bIdx = bh >> 6, xh = h >> 2;
  int tid = threadIdx.x, lane = tid & 31, wid = tid >> 5;
  int wm = wid >> 1, wn = wid & 1;
  int row0 = bIdx * 2048 + cIdx * 128;

  float Ah = -__expf(A_log[h]);
  if (tid < 128) sdt[tid] = dtbuf[(size_t)(row0 + tid) * 64 + h];
  __syncthreads();
  if (tid < 128) sdAc[tid] = sdt[tid] * Ah;
  __syncthreads();
  for (int off = 1; off < 128; off <<= 1) {  // Hillis-Steele cumsum
    float v = 0.f;
    if (tid < 128 && tid >= off) v = sdAc[tid - off];
    __syncthreads();
    if (tid < 128) sdAc[tid] += v;
    __syncthreads();
  }

#pragma unroll
  for (int i = 0; i < 32; ++i) {  // load x,B,C tiles (128x64 each)
    int idx = tid + i * 256;
    int t = idx >> 6, e = idx & 63;
    size_t grow = (size_t)(row0 + t) * 6144;
    float xv = xBC[grow + xh * 64 + e];
    float Bv = xBC[grow + 1024 + xh * 64 + e];
    float Cv = xBC[grow + 2048 + h * 64 + e];
    sXT[e * 128 + t] = tobf(xv * sdt[t]);
    sB[t * 64 + e] = tobf(Bv);
    sC[t * 64 + e] = tobf(Cv);
  }
  __syncthreads();

  // G = C @ B^T : 128x128x64, waves 4(M) x 2(N=64 each)
  v8f g[2][4];
#pragma unroll
  for (int i = 0; i < 2; ++i)
#pragma unroll
    for (int j = 0; j < 4; ++j) g[i][j] = vzero8();
  for (int k0 = 0; k0 < 64; k0 += 32) {
#pragma unroll
    for (int i = 0; i < 2; ++i) {
      v16bf a = frag_a(sC + (wm * 32 + i * 16) * 64 + k0, 64);
#pragma unroll
      for (int j = 0; j < 4; ++j) {
        v16bf b = frag_b(sB + (wn * 64 + j * 16) * 64 + k0, 64);
        g[i][j] = wmma_bf16(a, b, g[i][j]);
      }
    }
  }

  // Chunk states (VALU): states[p,n] = sum_t B[t,n]*exp(dAc_last-dAc[t])*xdt[t,p]
  float dAl = sdAc[127];
  {
    int p = tid & 63;
    int nb = (tid >> 6) << 4;
    float accs[16];
#pragma unroll
    for (int i = 0; i < 16; ++i) accs[i] = 0.f;
    for (int t = 0; t < 128; ++t) {
      float xd = (float)sXT[p * 128 + t] * __expf(dAl - sdAc[t]);
#pragma unroll
      for (int i = 0; i < 16; ++i) accs[i] += xd * (float)sB[t * 64 + nb + i];
    }
    size_t sbase = ((size_t)(bh * 32 + cIdx) * 64 + p) * 64 + nb;
#pragma unroll
    for (int i = 0; i < 16; ++i) states[sbase + i] = accs[i];
  }
  if (tid == 0) cdecay[bh * 32 + cIdx] = __expf(dAl);
  if (tid < 128) dAc_out[(size_t)(bh * 32 + cIdx) * 128 + tid] = sdAc[tid];
  __syncthreads();

  // masked/decayed G -> sG (bf16), overlaying sC/sB
  {
    int rlo = (lane >> 4) << 3, colb = lane & 15;
#pragma unroll
    for (int i = 0; i < 2; ++i)
#pragma unroll
      for (int j = 0; j < 4; ++j) {
        int mb = wm * 32 + i * 16 + rlo;
        int nb = wn * 64 + j * 16 + colb;
#pragma unroll
        for (int v = 0; v < 8; ++v) {
          int mi = mb + v;
          float val = (mi >= nb) ? g[i][j][v] * __expf(sdAc[mi] - sdAc[nb]) : 0.f;
          sG[mi * 128 + nb] = tobf(val);
        }
      }
  }
  __syncthreads();

  // Y_diag = sG[128][128] @ xdt : B-operand = sXT ([p][t] = [N][K])
  v8f ya[2][2];
#pragma unroll
  for (int i = 0; i < 2; ++i)
#pragma unroll
    for (int j = 0; j < 2; ++j) ya[i][j] = vzero8();
  for (int k0 = 0; k0 < 128; k0 += 32) {
#pragma unroll
    for (int i = 0; i < 2; ++i) {
      v16bf a = frag_a(sG + (wm * 32 + i * 16) * 128 + k0, 128);
#pragma unroll
      for (int j = 0; j < 2; ++j) {
        v16bf b = frag_b(sXT + (wn * 32 + j * 16) * 128 + k0, 128);
        ya[i][j] = wmma_bf16(a, b, ya[i][j]);
      }
    }
  }
  {
    int rlo = (lane >> 4) << 3, colb = lane & 15;
#pragma unroll
    for (int i = 0; i < 2; ++i)
#pragma unroll
      for (int j = 0; j < 2; ++j) {
        int tb = wm * 32 + i * 16 + rlo;
        int p = wn * 32 + j * 16 + colb;
#pragma unroll
        for (int v = 0; v < 8; ++v) {
          int t = tb + v;
          y[((size_t)(row0 + t) * 64 + h) * 64 + p] = ya[i][j][v];
        }
      }
  }
}

// ---------------------------------------------------------------------------
// SSD-B: sequential inter-chunk scan. prevs[c] = state entering chunk c.
// ---------------------------------------------------------------------------
__global__ __launch_bounds__(256) void ssd_scan_kernel(
    const float* __restrict__ states, const float* __restrict__ cdecay,
    float* __restrict__ prevs) {
  int bh = blockIdx.x;
  int e0 = threadIdx.x;
  float carry[16];
#pragma unroll
  for (int i = 0; i < 16; ++i) carry[i] = 0.f;
  for (int c = 0; c < 32; ++c) {
    float cd = cdecay[bh * 32 + c];
    size_t base = (size_t)(bh * 32 + c) * 4096;
#pragma unroll
    for (int i = 0; i < 16; ++i) {
      size_t idx = base + e0 + i * 256;
      float pv = carry[i];
      prevs[idx] = pv;
      carry[i] = states[idx] + cd * pv;
    }
  }
}

// ---------------------------------------------------------------------------
// SSD-C: Y_off = (C o exp(dAc)) @ prevs^T via WMMA; y += Y_off + D[h]*x.
// Dynamic LDS: sCe[128][64] | sP[64][64] | sdAc[128]
// ---------------------------------------------------------------------------
__global__ __launch_bounds__(256) void ssd_c_kernel(
    const float* __restrict__ xBC, const float* __restrict__ dAc_in,
    const float* __restrict__ prevs, const float* __restrict__ Dvec,
    float* __restrict__ y) {
  extern __shared__ char smem[];
  __bf16* sCe = (__bf16*)smem;             // [128][64]
  __bf16* sP  = (__bf16*)(smem + 16384);   // [64][64] ([p][n] = [N][K])
  float* sdAc = (float*)(smem + 16384 + 8192);
  int cIdx = blockIdx.x, bh = blockIdx.y;
  int h = bh & 63, bIdx = bh >> 6, xh = h >> 2;
  int tid = threadIdx.x, lane = tid & 31, wid = tid >> 5;
  int wm = wid >> 1, wn = wid & 1;
  int row0 = bIdx * 2048 + cIdx * 128;

  if (tid < 128) sdAc[tid] = dAc_in[(size_t)(bh * 32 + cIdx) * 128 + tid];
  __syncthreads();
#pragma unroll
  for (int i = 0; i < 32; ++i) {
    int idx = tid + i * 256;
    int t = idx >> 6, n = idx & 63;
    float Cv = xBC[(size_t)(row0 + t) * 6144 + 2048 + h * 64 + n];
    sCe[t * 64 + n] = tobf(Cv * __expf(sdAc[t]));
  }
#pragma unroll
  for (int i = 0; i < 16; ++i) {
    int idx = tid + i * 256;
    sP[idx] = tobf(prevs[(size_t)(bh * 32 + cIdx) * 4096 + idx]);
  }
  __syncthreads();

  v8f acc[2][2];
#pragma unroll
  for (int i = 0; i < 2; ++i)
#pragma unroll
    for (int j = 0; j < 2; ++j) acc[i][j] = vzero8();
  for (int k0 = 0; k0 < 64; k0 += 32) {
#pragma unroll
    for (int i = 0; i < 2; ++i) {
      v16bf a = frag_a(sCe + (wm * 32 + i * 16) * 64 + k0, 64);
#pragma unroll
      for (int j = 0; j < 2; ++j) {
        v16bf b = frag_b(sP + (wn * 32 + j * 16) * 64 + k0, 64);
        acc[i][j] = wmma_bf16(a, b, acc[i][j]);
      }
    }
  }
  float Dh = Dvec[h];
  int rlo = (lane >> 4) << 3, colb = lane & 15;
#pragma unroll
  for (int i = 0; i < 2; ++i)
#pragma unroll
    for (int j = 0; j < 2; ++j) {
      int tb = wm * 32 + i * 16 + rlo;
      int p = wn * 32 + j * 16 + colb;
#pragma unroll
      for (int v = 0; v < 8; ++v) {
        int t = tb + v;
        size_t yi = ((size_t)(row0 + t) * 64 + h) * 64 + p;
        float xv = xBC[(size_t)(row0 + t) * 6144 + xh * 64 + p];
        y[yi] += acc[i][j][v] + Dh * xv;
      }
    }
}

// ---------------------------------------------------------------------------
// Host launcher
// ---------------------------------------------------------------------------
extern "C" void kernel_launch(void* const* d_in, const int* in_sizes, int n_in,
                              void* d_out, int out_size, void* d_ws, size_t ws_size,
                              hipStream_t stream) {
  (void)in_sizes; (void)n_in; (void)out_size; (void)ws_size;
  const float* hs        = (const float*)d_in[0];
  const float* in_proj_w = (const float*)d_in[1];
  const float* conv_w    = (const float*)d_in[2];
  const float* conv_b    = (const float*)d_in[3];
  const float* dt_bias   = (const float*)d_in[4];
  const float* A_log     = (const float*)d_in[5];
  const float* Dvec      = (const float*)d_in[6];
  const float* norm_w    = (const float*)d_in[7];
  const float* out_proj_w= (const float*)d_in[8];
  const float* ln1_w     = (const float*)d_in[9];
  const float* ln2_w     = (const float*)d_in[10];
  const float* gate_w    = (const float*)d_in[11];
  const float* up_w      = (const float*)d_in[12];
  const float* down_w    = (const float*)d_in[13];
  float* out = (float*)d_out;
  float* ws = (float*)d_ws;

  // Workspace layout (floats). g overlays zxbcdt; gu overlays y+y2.
  const size_t WS_H12  = 0;                        // 8,388,608 (h1 / h2)
  const size_t WS_ZX   = WS_H12 + 8388608ull;      // 42,205,184 (zxbcdt; later g)
  const size_t WS_XBC  = WS_ZX + 42205184ull;      // 25,165,824
  const size_t WS_DT   = WS_XBC + 25165824ull;     // 262,144
  const size_t WS_DAC  = WS_DT + 262144ull;        // 524,288
  const size_t WS_CD   = WS_DAC + 524288ull;       // 4,096
  const size_t WS_ST   = WS_CD + 4096ull;          // 16,777,216
  const size_t WS_PREV = WS_ST + 16777216ull;      // 16,777,216
  const size_t WS_Y    = WS_PREV + 16777216ull;    // 16,777,216
  const size_t WS_Y2   = WS_Y + 16777216ull;       // 16,777,216
  const size_t WS_HMID = WS_Y2 + 16777216ull;      // 8,388,608
  float* h1    = ws + WS_H12;
  float* zx    = ws + WS_ZX;
  float* xBC   = ws + WS_XBC;
  float* dt    = ws + WS_DT;
  float* dAc   = ws + WS_DAC;
  float* cd    = ws + WS_CD;
  float* st    = ws + WS_ST;
  float* prevs = ws + WS_PREV;
  float* ybuf  = ws + WS_Y;
  float* y2    = ws + WS_Y2;
  float* hmid  = ws + WS_HMID;
  float* gbuf  = zx;          // reuse after z consumed
  float* gu    = ybuf;        // reuse y+y2 region (33,554,432 floats)
  float* h2    = h1;          // reuse after in_proj done

  const int M = 4096;  // b*l

  // 1. ln1
  rmsnorm_kernel<<<M, 256, 0, stream>>>(hs, ln1_w, h1, 2048);
  // 2. in_proj GEMM: [4096,2048] @ [2048,10304]
  gemm_kernel<0><<<dim3(81, 32), 256, 0, stream>>>(h1, in_proj_w, zx, nullptr,
                                                   M, 10304, 2048);
  // 3. dt = softplus(dt + bias)
  dt_kernel<<<1024, 256, 0, stream>>>(zx, dt_bias, dt);
  // 4. causal depthwise conv + SiLU -> xBC
  conv_kernel<<<dim3(24, M), 256, 0, stream>>>(zx, conv_w, conv_b, xBC);
  // 5. SSD per-chunk (WMMA): Y_diag, states, chunk decay, dAc
  ssd_a_kernel<<<dim3(32, 128), 256, 50176, stream>>>(xBC, dt, A_log, ybuf,
                                                      st, cd, dAc);
  // 6. inter-chunk scan
  ssd_scan_kernel<<<128, 256, 0, stream>>>(st, cd, prevs);
  // 7. Y_off + D*x (WMMA)
  ssd_c_kernel<<<dim3(32, 128), 256, 25088, stream>>>(xBC, dAc, prevs, Dvec, ybuf);
  // 8. gated rmsnorm: y2 = rmsnorm(y * silu(z)) * norm_w
  gated_norm_kernel<<<M, 256, 0, stream>>>(ybuf, zx, norm_w, y2);
  // 9. out_proj + residual: hmid = y2 @ out_proj_w + hs
  gemm_kernel<1><<<dim3(16, 32), 256, 0, stream>>>(y2, out_proj_w, hmid, hs,
                                                   M, 2048, 4096);
  // 10. ln2
  rmsnorm_kernel<<<M, 256, 0, stream>>>(hmid, ln2_w, h2, 2048);
  // 11. gate: g = silu(h2 @ gate_w)
  gemm_kernel<2><<<dim3(64, 32), 256, 0, stream>>>(h2, gate_w, gbuf, nullptr,
                                                   M, 8192, 2048);
  // 12. up: gu = (h2 @ up_w) * g
  gemm_kernel<3><<<dim3(64, 32), 256, 0, stream>>>(h2, up_w, gu, gbuf,
                                                   M, 8192, 2048);
  // 13. down + residual: out = gu @ down_w + hmid
  gemm_kernel<1><<<dim3(16, 32), 256, 0, stream>>>(gu, down_w, out, hmid,
                                                   M, 2048, 8192);
}
